// LinearAttention_v2_10608569221191
// MI455X (gfx1250) — matrix operand order repaired and checked by reading, not compiled, not run on hardware
//
#include <hip/hip_runtime.h>

// ---------------------------------------------------------------------------
// Types for CDNA5 WMMA (wave32): v_wmma_f32_16x16x32_bf16
// ---------------------------------------------------------------------------
typedef __attribute__((ext_vector_type(16))) __bf16       bf16x16;
typedef __attribute__((ext_vector_type(8)))  float        f32x8;
typedef __attribute__((ext_vector_type(4)))  unsigned int u32x4;
typedef __attribute__((ext_vector_type(4)))  float        f32x4;
typedef __attribute__((ext_vector_type(8)))  int          i32x8;
typedef __attribute__((ext_vector_type(4)))  int          i32x4;

union ABFrag {
    bf16x16 v;
    u32x4   u[2];
};

__device__ __forceinline__ unsigned short f2bf(float f) {
    unsigned int u = __float_as_uint(f);
    u += 0x7FFFu + ((u >> 16) & 1u);      // round-to-nearest-even
    return (unsigned short)(u >> 16);
}
__device__ __forceinline__ float bf2f(unsigned short h) {
    return __uint_as_float(((unsigned int)h) << 16);
}

// Use the Tensor Data Mover when the toolchain exposes it.
// This toolchain exposes the 6-arg form:
//   (uint32x4 g0, int32x8 g1, int32x4 g2, int32x4 g3, int32x8 extra, i32 cpol)
#if defined(__gfx1250__) && __has_builtin(__builtin_amdgcn_tensor_load_to_lds) && \
    __has_builtin(__builtin_amdgcn_s_wait_tensorcnt)
#define USE_TDM 1
#else
#define USE_TDM 0
#endif

#if USE_TDM
// LDS byte offset (relative to workgroup LDS base) of a generic pointer into
// a __shared__ array: generic -> addrspace(3) -> ptrtoint.
__device__ __forceinline__ unsigned lds_byte_off(const void* p) {
    return (unsigned)(unsigned long long)(__attribute__((address_space(3))) const char*)p;
}
#endif

// ---------------------------------------------------------------------------
// fp32 -> bf16 conversion (vectorized, grid-stride)
// ---------------------------------------------------------------------------
__global__ __launch_bounds__(256) void cvt_f32_to_bf16(
    const float* __restrict__ in, unsigned short* __restrict__ out, long long n4)
{
    long long i      = (long long)blockIdx.x * blockDim.x + threadIdx.x;
    long long stride = (long long)gridDim.x * blockDim.x;
    for (; i < n4; i += stride) {
        f32x4 x = ((const f32x4*)in)[i];
        unsigned long long p =
             (unsigned long long)f2bf(x.x)
           | ((unsigned long long)f2bf(x.y) << 16)
           | ((unsigned long long)f2bf(x.z) << 32)
           | ((unsigned long long)f2bf(x.w) << 48);
        ((unsigned long long*)out)[i] = p;
    }
}

// ---------------------------------------------------------------------------
// WMMA GEMM:  C[M x N] = A[M x K] * B[N x K]^T + bias[N]
// A, B are bf16 row-major (K contiguous).  Output bf16 or f32.
// Block = 256 threads = 8 waves, each wave owns a 64x64 tile (4x4 WMMA tiles),
// workgroup tile = 128 (M) x 256 (N).  Requires M%128==0, N%256==0, K%32==0.
// ---------------------------------------------------------------------------
template <int OUT_BF16>
__global__ __launch_bounds__(256) void gemm_bf16_wmma(
    const unsigned short* __restrict__ A,
    const unsigned short* __restrict__ B,
    const float* __restrict__ bias,
    void* __restrict__ C,
    int M, int N, int K)
{
    const int lane = threadIdx.x & 31;
    const int wave = threadIdx.x >> 5;
    const int l16  = lane & 15;
    const int lh   = lane >> 4;                 // half-wave: 0 or 1

    const int m0 = blockIdx.y * 128 + (wave >> 2) * 64;
    const int n0 = blockIdx.x * 256 + (wave & 3) * 64;

    // A fragment rows: lane holds row l16, K-chunks at lh*8 and lh*8+16
    const unsigned short* aptr[4];
    // B fragment rows: lane holds column (= weight row) l16, K-chunks at lh*8 and lh*8+16
    const unsigned short* bptr[4];
#pragma unroll
    for (int i = 0; i < 4; ++i)
        aptr[i] = A + (long long)(m0 + 16 * i + l16) * K + lh * 8;
#pragma unroll
    for (int j = 0; j < 4; ++j)
        bptr[j] = B + (long long)(n0 + 16 * j + l16) * K + lh * 8;

    f32x8 acc[4][4] = {};

    for (int k = 0; k < K; k += 32) {
        ABFrag a[4], b[4];
#pragma unroll
        for (int i = 0; i < 4; ++i) {
            a[i].u[0] = *(const u32x4*)(aptr[i] + k);
            a[i].u[1] = *(const u32x4*)(aptr[i] + k + 16);
        }
#pragma unroll
        for (int j = 0; j < 4; ++j) {
            b[j].u[0] = *(const u32x4*)(bptr[j] + k);
            b[j].u[1] = *(const u32x4*)(bptr[j] + k + 16);
        }
        // prefetch a few K-steps ahead (global_prefetch_b8)
        if (k + 256 < K) {
#pragma unroll
            for (int i = 0; i < 4; ++i) __builtin_prefetch(aptr[i] + k + 256, 0, 3);
#pragma unroll
            for (int j = 0; j < 4; ++j) __builtin_prefetch(bptr[j] + k + 256, 0, 3);
        }
#pragma unroll
        for (int i = 0; i < 4; ++i)
#pragma unroll
            for (int j = 0; j < 4; ++j)
                acc[i][j] = __builtin_amdgcn_wmma_f32_16x16x32_bf16(
                    false, a[i].v, false, b[j].v, (short)0, acc[i][j], false, false);
    }

    // Epilogue: D layout -> lane l16 owns column n0+16j+l16, rows r + 8*lh.
    float bv[4];
#pragma unroll
    for (int j = 0; j < 4; ++j) bv[j] = bias[n0 + 16 * j + l16];

#pragma unroll
    for (int i = 0; i < 4; ++i) {
#pragma unroll
        for (int j = 0; j < 4; ++j) {
            const long long col = n0 + 16 * j + l16;
#pragma unroll
            for (int r = 0; r < 8; ++r) {
                const long long row = m0 + 16 * i + lh * 8 + r;
                const float val = acc[i][j][r] + bv[j];
                if (OUT_BF16)
                    ((unsigned short*)C)[row * N + col] = f2bf(val);
                else
                    ((float*)C)[row * N + col] = val;
            }
        }
    }
}

// ---------------------------------------------------------------------------
// Per-token linear attention.  One wave per token; token qkv row (6 KB bf16)
// is DMA'd into LDS by the Tensor Data Mover, then:
//   k~ = softmax over h (per column d)   [lane owns cols d=lane, lane+32],
//        written back in place (bf16) over the k region of the LDS tile
//   ctx[d][e] = sum_h k~[h][d] * v[h][e] [lane owns cols e=lane, lane+32]
//   out[h][e] = scale * sum_d q[h][d] * ctx[d][e]
// Output written (B,H,N,hd)-contiguous so the proj GEMM reads it flat.
// ---------------------------------------------------------------------------
__global__ __launch_bounds__(256) void linattn_kernel(
    const unsigned short* __restrict__ qkv,
    unsigned short* __restrict__ attn,
    int Nseq)
{
    __shared__ __align__(16) unsigned char smem[8 * 6144];   // 48 KB: 6 KB/wave
    const int lane = threadIdx.x & 31;
    const int wave = threadIdx.x >> 5;
    const long long t = (long long)blockIdx.x * 8 + wave;

    unsigned char* wbase = smem + wave * 6144;

    // ---- stage this wave's token row (3072 bf16 = 6144 B) into LDS ----
#if USE_TDM
    {
        const unsigned long long ga = (unsigned long long)(const void*)(qkv + t * 3072);
        const unsigned int       la = lds_byte_off(wbase);
        // D# group 0: count=1 (user), lds_addr, global_addr[56:0], type=2
        u32x4 g0;
        g0.x = 1u;
        g0.y = la;
        g0.z = (unsigned int)ga;
        g0.w = ((unsigned int)(ga >> 32) & 0x01FFFFFFu) | 0x80000000u;
        // D# group 1: data_size=1 (2B), tensor_dim0=3072, tensor_dim1=1,
        //             tile_dim0=3072, tile_dim1=1, dim0_stride=3072
        i32x8 g1;
        g1[0] = 0x00010000;          // [17:16] data_size = 1 (2 bytes)
        g1[1] = (int)(3072u << 16);  // [63:48] tensor_dim0 lo16
        g1[2] = 0x00010000;          // [79:64] dim0 hi16 = 0, [95:80] tensor_dim1 = 1
        g1[3] = (int)(3072u << 16);  // [127:112] tile_dim0 = 3072
        g1[4] = 1;                   // [143:128] tile_dim1 = 1, [159:144] tile_dim2 = 0
        g1[5] = 3072;                // [191:160] tensor_dim0_stride lo32
        g1[6] = (int)(3072u << 16);  // [207:192] stride0 hi = 0, [223:208] stride1 lo = 3072
        g1[7] = 0;
        const i32x4 z4 = {0, 0, 0, 0};
        const i32x8 z8 = {0, 0, 0, 0, 0, 0, 0, 0};
        __builtin_amdgcn_tensor_load_to_lds(g0, g1, z4, z4, z8, 0);
        __builtin_amdgcn_s_wait_tensorcnt((short)0);
    }
#else
    {
        const unsigned short* g = qkv + t * 3072;
#pragma unroll
        for (int c = 0; c < 12; ++c) {
            const int chunk = c * 32 + lane;                 // 384 x 16B chunks
            *(u32x4*)(wbase + chunk * 16) = *(const u32x4*)(g + chunk * 8);
        }
    }
#endif

    const unsigned short* tokq = (const unsigned short*)wbase;        // q: [0,1024)
    unsigned short*       kbuf = (unsigned short*)wbase + 1024;       // k: [1024,2048)
    const unsigned short* vbuf = (const unsigned short*)wbase + 2048; // v: [2048,3072)

    // --- softmax(k) over heads, two columns per lane; k~ written in place ---
    {
        float e0[16], e1[16];
        float s0 = 0.f, s1 = 0.f;
#pragma unroll
        for (int h = 0; h < 16; ++h) {
            e0[h] = __expf(bf2f(kbuf[h * 64 + lane]));
            e1[h] = __expf(bf2f(kbuf[h * 64 + lane + 32]));
            s0 += e0[h];
            s1 += e1[h];
        }
        const float r0 = 1.0f / s0, r1 = 1.0f / s1;
#pragma unroll
        for (int h = 0; h < 16; ++h) {
            kbuf[h * 64 + lane]      = f2bf(e0[h] * r0);
            kbuf[h * 64 + lane + 32] = f2bf(e1[h] * r1);
        }
    }
    __syncthreads();

    // --- context = k~^T v : two columns of ctx per lane, kept in registers ---
    float c0[64] = {}, c1[64] = {};
    for (int h = 0; h < 16; ++h) {
        const float v0 = bf2f(vbuf[h * 64 + lane]);
        const float v1 = bf2f(vbuf[h * 64 + lane + 32]);
        const unsigned short* kr = kbuf + h * 64;   // LDS row broadcast
#pragma unroll
        for (int d = 0; d < 64; ++d) {
            const float kv = bf2f(kr[d]);
            c0[d] = fmaf(kv, v0, c0[d]);
            c1[d] = fmaf(kv, v1, c1[d]);
        }
    }

    // --- out = scale * q @ ctx, streamed row by row ---
    const long long b = t / Nseq;
    const long long n = t % Nseq;
    unsigned short* obase = attn + ((b * 16) * (long long)Nseq + n) * 64;
    const float scale = 0.125f;                 // 64^-0.5
    for (int h = 0; h < 16; ++h) {
        const unsigned short* qrow = tokq + h * 64;   // wave-uniform LDS row
        float a0 = 0.f, a1 = 0.f;
#pragma unroll
        for (int d = 0; d < 64; ++d) {
            const float qd = bf2f(qrow[d]);
            a0 = fmaf(qd, c0[d], a0);
            a1 = fmaf(qd, c1[d], a1);
        }
        unsigned short* orow = obase + (long long)h * Nseq * 64;
        orow[lane]      = f2bf(a0 * scale);
        orow[lane + 32] = f2bf(a1 * scale);
    }
}

// ---------------------------------------------------------------------------
// Host-side orchestration
// ---------------------------------------------------------------------------
extern "C" void kernel_launch(void* const* d_in, const int* in_sizes, int n_in,
                              void* d_out, int out_size, void* d_ws, size_t ws_size,
                              hipStream_t stream)
{
    (void)in_sizes; (void)n_in; (void)out_size; (void)ws_size;

    const float* x      = (const float*)d_in[0];   // (8,4096,1024)
    const float* w_qkv  = (const float*)d_in[1];   // (3072,1024)
    const float* b_qkv  = (const float*)d_in[2];   // (3072)
    const float* w_proj = (const float*)d_in[3];   // (1024,1024)
    const float* b_proj = (const float*)d_in[4];   // (1024)

    const int       Nseq  = 4096;
    const int       Cdim  = 1024;
    const long long M     = 8LL * Nseq;            // 32768 tokens
    const int       Nqkv  = 3 * Cdim;              // 3072

    // Workspace layout (bytes, 256-aligned):
    char* ws = (char*)d_ws;
    unsigned short* xb     = (unsigned short*)(ws);                      //  64 MB
    unsigned short* wqkvb  = (unsigned short*)(ws + 67108864ULL);        //   6 MB
    unsigned short* wprojb = (unsigned short*)(ws + 73400320ULL);        //   2 MB
    unsigned short* qkvb   = (unsigned short*)(ws + 75497472ULL);        // 192 MB
    unsigned short* attnb  = (unsigned short*)(ws + 276824064ULL);       //  64 MB

    // 1) fp32 -> bf16 conversions
    {
        long long n4 = M * Cdim / 4;             // 8,388,608
        cvt_f32_to_bf16<<<(unsigned)((n4 + 255) / 256), 256, 0, stream>>>(x, xb, n4);
    }
    {
        long long n4 = (long long)Nqkv * Cdim / 4;
        cvt_f32_to_bf16<<<(unsigned)((n4 + 255) / 256), 256, 0, stream>>>(w_qkv, wqkvb, n4);
    }
    {
        long long n4 = (long long)Cdim * Cdim / 4;
        cvt_f32_to_bf16<<<(unsigned)((n4 + 255) / 256), 256, 0, stream>>>(w_proj, wprojb, n4);
    }

    // 2) QKV GEMM: (32768 x 1024) x (1024 x 3072) + b_qkv -> bf16
    {
        dim3 grid(Nqkv / 256, (unsigned)(M / 128));
        gemm_bf16_wmma<1><<<grid, 256, 0, stream>>>(xb, wqkvb, b_qkv, qkvb,
                                                    (int)M, Nqkv, Cdim);
    }

    // 3) Per-token linear attention -> (B,H,N,hd) bf16 (== proj-GEMM input rows)
    linattn_kernel<<<(unsigned)(M / 8), 256, 0, stream>>>(qkvb, attnb, Nseq);

    // 4) Projection GEMM: (32768 x 1024) x (1024 x 1024) + b_proj -> f32 d_out
    {
        dim3 grid(Cdim / 256, (unsigned)(M / 128));
        gemm_bf16_wmma<0><<<grid, 256, 0, stream>>>(attnb, wprojb, b_proj, d_out,
                                                    (int)M, Cdim, Cdim);
    }
}
